// Qwen3MLP_49890340110676
// MI455X (gfx1250) — matrix operand order repaired and checked by reading, not compiled, not run on hardware
//
#include <hip/hip_runtime.h>
#include <hip/hip_bf16.h>

#define HID   4096
#define INTER 12288
#define TOK   8192
#define GS    128

typedef __attribute__((ext_vector_type(16))) __bf16 v16bf;
typedef __attribute__((ext_vector_type(8)))  __bf16 v8bf;
typedef __attribute__((ext_vector_type(8)))  float  v8f;
typedef int v4i __attribute__((vector_size(16)));

// ---- gfx1250 async global->LDS copy (ASYNCcnt path), guarded so compile
// ---- cannot regress if the builtin is absent on this toolchain.
#if __has_builtin(__builtin_amdgcn_global_load_async_to_lds_b128)
#define HAVE_ASYNC_LDS 1
static __device__ __forceinline__ void async_copy_b128(const void* gsrc, void* ldst) {
    __builtin_amdgcn_global_load_async_to_lds_b128(
        (__attribute__((address_space(1))) v4i*)(gsrc),
        (__attribute__((address_space(3))) v4i*)(ldst),
        /*imm offset=*/0, /*cpol=*/0);
}
static __device__ __forceinline__ void wait_async0() {
#if __has_builtin(__builtin_amdgcn_s_wait_asynccnt)
    __builtin_amdgcn_s_wait_asynccnt(0);
#else
    asm volatile("s_wait_asynccnt 0x0" ::: "memory");
#endif
}
#endif

// float -> bf16 (round-to-nearest-even), bit-level.
static __device__ __forceinline__ unsigned short f2bf(float f) {
    union { float f; unsigned u; } x; x.f = f;
    unsigned r = x.u + 0x7FFFu + ((x.u >> 16) & 1u);
    return (unsigned short)(r >> 16);
}

// Unpack 8 int4 nibbles (K-major) of one packed word, dequant, emit 8 bf16.
static __device__ __forceinline__ void dequant8(int w, int z, float s, unsigned short* out) {
#pragma unroll
    for (int j = 0; j < 8; ++j) {
        int q = (w >> (4 * j)) & 0xF;
        out[j] = f2bf((float)(q - z) * s);
    }
}

// Dequantize a 64(K) x 64(N) weight tile into LDS, stored [n][k] contiguous-in-k
// (pitch 72 halves) so B fragments are contiguous ds_load_b128 pairs.
static __device__ __forceinline__ void load_wtile(const int* __restrict__ qw,
                                                  const int* __restrict__ qz,
                                                  const float* __restrict__ sc,
                                                  int N, int kb, int n0,
                                                  unsigned short* Bl, int tid) {
    const int g = kb / GS;  // whole 64-K block sits in one quant group
#pragma unroll
    for (int i = 0; i < 2; ++i) {
        int idx  = tid * 2 + i;       // 0..511 over 8 rows x 64 cols
        int row8 = idx >> 6;          // packed-int32 row within tile (k = row8*8+j)
        int n    = idx & 63;
        int gn   = n0 + n;
        int w    = qw[(kb / 8 + row8) * N + gn];
        int zc   = qz[g * (N / 8) + (gn >> 3)];
        int z    = (zc >> (4 * (gn & 7))) & 0xF;
        float s  = sc[g * N + gn];
        union { v8bf v; unsigned short u[8]; } d;
        dequant8(w, z, s, d.u);
        *(v8bf*)&Bl[n * 72 + row8 * 8] = d.v;  // 16B-aligned (144*n + 16*row8)
    }
}

// A fragment (16x32 bf16, MxK): lane holds m = base_m + lane%16,
// k = 16*hi .. 16*hi+15 (contiguous), hi = lane/16.
static __device__ __forceinline__ v16bf load_a_frag(const unsigned short* Al,
                                                    int m, int k0, int lane) {
    union { v16bf v; v8bf h[2]; } f;
    int khalf = (lane >> 4) * 16;
    const unsigned short* p = &Al[m * 72 + k0 + khalf];
    f.h[0] = *(const v8bf*)p;
    f.h[1] = *(const v8bf*)(p + 8);
    return f.v;
}

// B fragment (32x16 bf16, KxN): lane holds n = base_n + lane%16,
// k = 16*(lane/16) .. +15 (16 contiguous halves in [n][k] layout).
static __device__ __forceinline__ v16bf load_b_frag(const unsigned short* Bl,
                                                    int n, int k0, int lane) {
    union { v16bf v; v8bf h[2]; } f;
    int kb16 = (lane >> 4) * 16;
    const unsigned short* p = &Bl[n * 72 + k0 + kb16];
    f.h[0] = *(const v8bf*)p;
    f.h[1] = *(const v8bf*)(p + 8);
    return f.v;
}

// ---------------- Kernel 1: h = silu(x @ Wg) * (x @ Wu), h in bf16 ------------
__global__ __launch_bounds__(256)
void gateup_kernel(const float* __restrict__ x,
                   const int* __restrict__ gq, const int* __restrict__ gz,
                   const float* __restrict__ gs,
                   const int* __restrict__ uq, const int* __restrict__ uz,
                   const float* __restrict__ us,
                   unsigned short* __restrict__ h) {
    __shared__ unsigned short Ax[128 * 72];
    __shared__ unsigned short Bg[64 * 72];
    __shared__ unsigned short Bu[64 * 72];

    const int tid  = threadIdx.x;
    const int lane = tid & 31;
    const int wave = tid >> 5;
    const int waveM = wave & 3;   // 4 waves along M (32 rows each)
    const int waveN = wave >> 2;  // 2 waves along N (32 cols each)
    const int row0 = blockIdx.y * 128;
    const int n0   = blockIdx.x * 64;

    v8f accG[2][2] = {};
    v8f accU[2][2] = {};

    for (int kb = 0; kb < HID; kb += 64) {
        __syncthreads();
        // Stage x tile 128x64 fp32 -> bf16 LDS [m][k], pitch 72.
#pragma unroll
        for (int i = 0; i < 8; ++i) {
            int f  = tid + i * 256;          // 0..2047 float4s
            int r  = f >> 4;
            int c4 = f & 15;
            const float4 xv = *(const float4*)&x[(row0 + r) * HID + kb + c4 * 4];
            union { uint2 u2; unsigned short s[4]; } d;
            d.s[0] = f2bf(xv.x); d.s[1] = f2bf(xv.y);
            d.s[2] = f2bf(xv.z); d.s[3] = f2bf(xv.w);
            *(uint2*)&Ax[r * 72 + c4 * 4] = d.u2;
        }
        load_wtile(gq, gz, gs, INTER, kb, n0, Bg, tid);
        load_wtile(uq, uz, us, INTER, kb, n0, Bu, tid);
        if (kb + 64 < HID) {  // prefetch next K tiles (global_prefetch_b8)
            __builtin_prefetch(&x[(row0 + (tid >> 1)) * HID + kb + 64], 0, 1);
            __builtin_prefetch(&gq[((kb + 64) / 8 + (tid >> 6)) * INTER + n0 + (tid & 63)], 0, 1);
            __builtin_prefetch(&uq[((kb + 64) / 8 + (tid >> 6)) * INTER + n0 + (tid & 63)], 0, 1);
        }
        __syncthreads();

#pragma unroll
        for (int ks = 0; ks < 2; ++ks) {
            const int k0 = ks * 32;
            v16bf a[2], bgf[2], buf[2];
#pragma unroll
            for (int mt = 0; mt < 2; ++mt)
                a[mt] = load_a_frag(Ax, waveM * 32 + mt * 16 + (lane & 15), k0, lane);
#pragma unroll
            for (int nt = 0; nt < 2; ++nt) {
                bgf[nt] = load_b_frag(Bg, waveN * 32 + nt * 16 + (lane & 15), k0, lane);
                buf[nt] = load_b_frag(Bu, waveN * 32 + nt * 16 + (lane & 15), k0, lane);
            }
#pragma unroll
            for (int mt = 0; mt < 2; ++mt)
#pragma unroll
                for (int nt = 0; nt < 2; ++nt) {
                    accG[mt][nt] = __builtin_amdgcn_wmma_f32_16x16x32_bf16(
                        false, a[mt], false, bgf[nt], (short)0, accG[mt][nt], false, false);
                    accU[mt][nt] = __builtin_amdgcn_wmma_f32_16x16x32_bf16(
                        false, a[mt], false, buf[nt], (short)0, accU[mt][nt], false, false);
                }
        }
    }

    // Epilogue: silu(g)*u -> bf16, staged through LDS for b128-coalesced stores.
    __syncthreads();  // Ax reuse: make sure all waves finished reading it
    const int nlane = lane & 15;
    const int mhi   = (lane >> 4) * 8;
#pragma unroll
    for (int mt = 0; mt < 2; ++mt)
#pragma unroll
        for (int nt = 0; nt < 2; ++nt)
#pragma unroll
            for (int v = 0; v < 8; ++v) {
                float g = accG[mt][nt][v];
                float u = accU[mt][nt][v];
                float val = (g / (1.0f + __expf(-g))) * u;
                int ml = waveM * 32 + mt * 16 + mhi + v;
                int nl = waveN * 32 + nt * 16 + nlane;
                Ax[ml * 72 + nl] = f2bf(val);
            }
    __syncthreads();
#pragma unroll
    for (int i = 0; i < 4; ++i) {
        int f  = tid + i * 256;          // 0..1023 uint4s over 128x64 halves
        int r  = f >> 3;
        int c8 = f & 7;
        uint4 hv = *(const uint4*)&Ax[r * 72 + c8 * 8];
        *(uint4*)&h[(size_t)(row0 + r) * INTER + n0 + c8 * 8] = hv;
    }
}

// ---------------- Kernel 2: out = h @ Wdown (f32 out) -------------------------
__global__ __launch_bounds__(256)
void down_kernel(const unsigned short* __restrict__ h,
                 const int* __restrict__ dq, const int* __restrict__ dz,
                 const float* __restrict__ dsc,
                 float* __restrict__ out) {
    __shared__ unsigned short Ax[128 * 72];
    __shared__ unsigned short Bd[64 * 72];

    const int tid  = threadIdx.x;
    const int lane = tid & 31;
    const int wave = tid >> 5;
    const int waveM = wave & 3;
    const int waveN = wave >> 2;
    const int row0 = blockIdx.y * 128;
    const int n0   = blockIdx.x * 64;

    v8f acc[2][2] = {};

    for (int kb = 0; kb < INTER; kb += 64) {
        __syncthreads();
        // Stage h tile 128x64 bf16: pure copy -> async DMA to LDS when available.
#pragma unroll
        for (int i = 0; i < 4; ++i) {
            int f  = tid + i * 256;          // 0..1023 uint4s (8 halves each)
            int r  = f >> 3;
            int c8 = f & 7;
#ifdef HAVE_ASYNC_LDS
            async_copy_b128(&h[(size_t)(row0 + r) * INTER + kb + c8 * 8],
                            &Ax[r * 72 + c8 * 8]);
#else
            uint4 hv = *(const uint4*)&h[(size_t)(row0 + r) * INTER + kb + c8 * 8];
            *(uint4*)&Ax[r * 72 + c8 * 8] = hv;
#endif
        }
        load_wtile(dq, dz, dsc, HID, kb, n0, Bd, tid);
        if (kb + 64 < INTER) {
            __builtin_prefetch(&h[(size_t)(row0 + (tid >> 1)) * INTER + kb + 64], 0, 1);
            __builtin_prefetch(&dq[((kb + 64) / 8 + (tid >> 6)) * HID + n0 + (tid & 63)], 0, 1);
        }
#ifdef HAVE_ASYNC_LDS
        wait_async0();   // my async LDS writes landed; barrier covers the rest
#endif
        __syncthreads();

#pragma unroll
        for (int ks = 0; ks < 2; ++ks) {
            const int k0 = ks * 32;
            v16bf a[2], b[2];
#pragma unroll
            for (int mt = 0; mt < 2; ++mt)
                a[mt] = load_a_frag(Ax, waveM * 32 + mt * 16 + (lane & 15), k0, lane);
#pragma unroll
            for (int nt = 0; nt < 2; ++nt)
                b[nt] = load_b_frag(Bd, waveN * 32 + nt * 16 + (lane & 15), k0, lane);
#pragma unroll
            for (int mt = 0; mt < 2; ++mt)
#pragma unroll
                for (int nt = 0; nt < 2; ++nt)
                    acc[mt][nt] = __builtin_amdgcn_wmma_f32_16x16x32_bf16(
                        false, a[mt], false, b[nt], (short)0, acc[mt][nt], false, false);
        }
    }

    const int nlane = lane & 15;
    const int mhi   = (lane >> 4) * 8;
#pragma unroll
    for (int mt = 0; mt < 2; ++mt)
#pragma unroll
        for (int nt = 0; nt < 2; ++nt)
#pragma unroll
            for (int v = 0; v < 8; ++v) {
                int m = row0 + waveM * 32 + mt * 16 + mhi + v;
                int n = n0 + waveN * 32 + nt * 16 + nlane;
                out[(size_t)m * HID + n] = acc[mt][nt][v];
            }
}

extern "C" void kernel_launch(void* const* d_in, const int* in_sizes, int n_in,
                              void* d_out, int out_size, void* d_ws, size_t ws_size,
                              hipStream_t stream) {
    (void)in_sizes; (void)n_in; (void)out_size; (void)ws_size;
    const float* x  = (const float*)d_in[0];
    const int*   gq = (const int*)d_in[1];
    const int*   gz = (const int*)d_in[2];
    const float* gs = (const float*)d_in[3];
    const int*   uq = (const int*)d_in[4];
    const int*   uz = (const int*)d_in[5];
    const float* us = (const float*)d_in[6];
    const int*   dq = (const int*)d_in[7];
    const int*   dz = (const int*)d_in[8];
    const float* dsc = (const float*)d_in[9];

    unsigned short* h = (unsigned short*)d_ws;  // [TOK, INTER] bf16 intermediate

    dim3 blk(256);
    dim3 grdA(INTER / 64, TOK / 128);  // 192 x 64
    gateup_kernel<<<grdA, blk, 0, stream>>>(x, gq, gz, gs, uq, uz, us, h);

    dim3 grdB(HID / 64, TOK / 128);    // 64 x 64
    down_kernel<<<grdB, blk, 0, stream>>>(h, dq, dz, dsc, (float*)d_out);
}
